// CTDTMemoryUpdater_9423158247662
// MI455X (gfx1250) — compile-verified, matrix operand order checked by reading, not run on hardware
//
#include <hip/hip_runtime.h>
#include <hip/hip_bf16.h>
#include <stdint.h>

// Problem constants (from reference)
#define N_NODES 100000
#define MDIM    256         // MEM_DIM == MSG_DIM == HID_DIM
#define BATCH   20000
#define DIVIDE  8
#define TILES_B (BATCH / 16)        // 1250 exact
#define WAVES_PER_BLOCK 8
#define THREADS (WAVES_PER_BLOCK * 32)
#define WSZ     (2 * MDIM * MDIM)   // elems in one [2M, M] weight = 131072

// LDS staging: one slot = 16 weight rows x 256 bf16, padded by TDM to a
// 528-byte row stride (512B data + 16B pad) = 33 uint4 per row.
#define ROW_U4   33
#define SLOT_U4  (16 * ROW_U4)          // 528 uint4 = 8448 bytes

typedef __attribute__((ext_vector_type(16))) __bf16 v16bf;
typedef __attribute__((ext_vector_type(8)))  float  v8f;
typedef unsigned int v4u __attribute__((ext_vector_type(4)));
typedef int          v8i __attribute__((ext_vector_type(8)));
typedef int          v4i __attribute__((ext_vector_type(4)));

extern __shared__ uint4 smem4[];   // dynamic LDS (addrspace(3) -> ds_load)

union FragAB {
  v16bf v;
  uint4 q[2];
  unsigned short e[16];
};

__device__ __forceinline__ unsigned short f2bf(float f) {
  unsigned u = __float_as_uint(f);
  u += 0x7FFFu + ((u >> 16) & 1u);   // round-to-nearest-even
  return (unsigned short)(u >> 16);
}
__device__ __forceinline__ float fsigmoid(float x) {
  return 1.0f / (1.0f + __expf(-x));
}
__device__ __forceinline__ float ftanh(float x) {
  float e = __expf(2.0f * x);
  return (e - 1.0f) / (e + 1.0f);
}

// ---- Tensor Data Mover: stage one 16x256 bf16 weight slice into LDS -------
// D# group0: count=1, lds byte address, 57-bit global address, type=2(image).
// D# group1: data_size=2B; pad_enable, pad_interval=128 DWORDs (one 512B row),
//            pad_amount=4 DWORDs (16B) -> LDS row stride 528B;
//            tensor_dim0=256, tensor_dim1=16, tile 256x16, dim0 stride 256.
// This toolchain's builtin takes 6 args: (v4u, v8i, v4i, v4i, v8i, i32 cpol).
__device__ __forceinline__ void tdm_load_tile(const unsigned short* W,
                                              int rowbase, unsigned lds_byte) {
  unsigned long long ga =
      (unsigned long long)(uintptr_t)(W + (size_t)rowbase * MDIM);
  v4u g0 = { 0x1u,                                  // count=1, user mode
             lds_byte,                               // lds_addr
             (unsigned)(ga & 0xFFFFFFFFu),           // global_addr[31:0]
             (unsigned)((ga >> 32) & 0x01FFFFFFu) |  // global_addr[56:32]
                 0x80000000u };                      // type=2 ("image")
  v8i g1 = { (int)0x07910000,        // data_size=1(2B), pad_en, int=128dw, amt=4dw
             (int)(MDIM << 16),      // tensor_dim0[15:0]=256 in [31:16]
             (int)(16 << 16),        // tensor_dim0 hi=0 | tensor_dim1=16
             (int)(MDIM << 16),      // tensor_dim1 hi=0 | tile_dim0=256
             16,                     // tile_dim1=16 | tile_dim2=0
             MDIM,                   // tensor_dim0_stride=256
             0, 0 };                 // stride0 hi | tensor_dim1_stride=0
  v4i g2 = { 0, 0, 0, 0 };
  v4i g3 = { 0, 0, 0, 0 };
  v8i g4 = { 0, 0, 0, 0, 0, 0, 0, 0 };
  __builtin_amdgcn_tensor_load_to_lds(g0, g1, g2, g3, g4, 0);
}

// Load one fp32 activation row into the 8 bf16 A-fragments (ISA 16-bit 16x32
// A layout: lane m=lane&15 holds row m; kh=lane>>4 selects the K-halves).
__device__ __forceinline__ void load_a_row(FragAB a[8],
                                           const float* __restrict__ row,
                                           int kh) {
#pragma unroll
  for (int k = 0; k < 8; ++k) {
    const float* p = row + k * 32 + kh * 8;
    float4 f0 = *(const float4*)(p);
    float4 f1 = *(const float4*)(p + 4);
    float4 f2 = *(const float4*)(p + 16);
    float4 f3 = *(const float4*)(p + 20);
    a[k].e[0]  = f2bf(f0.x); a[k].e[1]  = f2bf(f0.y);
    a[k].e[2]  = f2bf(f0.z); a[k].e[3]  = f2bf(f0.w);
    a[k].e[4]  = f2bf(f1.x); a[k].e[5]  = f2bf(f1.y);
    a[k].e[6]  = f2bf(f1.z); a[k].e[7]  = f2bf(f1.w);
    a[k].e[8]  = f2bf(f2.x); a[k].e[9]  = f2bf(f2.y);
    a[k].e[10] = f2bf(f2.z); a[k].e[11] = f2bf(f2.w);
    a[k].e[12] = f2bf(f3.x); a[k].e[13] = f2bf(f3.y);
    a[k].e[14] = f2bf(f3.z); a[k].e[15] = f2bf(f3.w);
  }
}

// One 16x16 output tile from an LDS-staged weight slice (slot base in uint4
// units). Per lane: feature row n = lane&15, K-halves via kh = lane>>4.
__device__ __forceinline__ v8f gemm_tile_lds(const FragAB a[8], int base_u4,
                                             int nbias,
                                             const float* __restrict__ bias,
                                             int lane) {
  const int n  = lane & 15;
  const int kh = lane >> 4;
  const float bv = bias[nbias + n];
  v8f acc = { bv, bv, bv, bv, bv, bv, bv, bv };
  const int row = base_u4 + n * ROW_U4 + kh;
#pragma unroll
  for (int k = 0; k < 8; ++k) {
    FragAB b;
    b.q[0] = smem4[row + k * 4];        // ds_load_b128
    b.q[1] = smem4[row + k * 4 + 2];
    acc = __builtin_amdgcn_wmma_f32_16x16x32_bf16(
        false, a[k].v, false, b.v, (short)0, acc, false, false);
  }
  return acc;
}

// Pack the three fp32 weight matrices to bf16 in workspace:
// [0, WSZ) = W_C, [WSZ, 2*WSZ) = W_D, [2*WSZ, 4*WSZ) = W_h
__global__ __launch_bounds__(256) void cvt_weights(
    const float* __restrict__ WC, const float* __restrict__ WD,
    const float* __restrict__ WH, unsigned short* __restrict__ dst) {
  int i = blockIdx.x * 256 + threadIdx.x;    // 4*WSZ total = 524288
  float v;
  if (i < WSZ)          v = WC[i];
  else if (i < 2 * WSZ) v = WD[i - WSZ];
  else                  v = WH[i - 2 * WSZ];
  dst[i] = f2bf(v);
}

template <bool FINAL>
__global__ __launch_bounds__(THREADS) void tgn_step(
    float* __restrict__ mem,                    // in-place memory (== d_out)
    const long long* __restrict__ ids,          // [BATCH] int64
    const float* __restrict__ msg,              // [BATCH, 256]
    const float* __restrict__ dtdg,             // [N_NODES, 256]
    const unsigned short* __restrict__ WCb, const float* __restrict__ bC,
    const unsigned short* __restrict__ WDb, const float* __restrict__ bD,
    const unsigned short* __restrict__ WHb, const float* __restrict__ bH) {
  const int lane = threadIdx.x & 31;
  const int wave = threadIdx.x >> 5;
  const int tile = blockIdx.x * WAVES_PER_BLOCK + wave;
  const bool active = (tile < TILES_B);   // wave-uniform; EXEC stays full
  const int rb = tile * 16;
  const int kh = lane >> 4;
  const int mA = lane & 15;

  constexpr int NSLOTS = FINAL ? 8 : 4;
  const bool issuer = (wave < NSLOTS);    // wave-uniform
  const unsigned lds_base = (unsigned)(uintptr_t)&smem4[0];

  // Staging schedule: slot s of an iteration t (output cols nb = t*16) holds
  //   s=0: W_C[nb]      s=1: W_C[M+nb]   s=2: W_h[M+nb]   s=3: W_h[3M+nb]
  //   s=4: W_D[nb]      s=5: W_D[M+nb]   s=6: W_h[nb]     s=7: W_h[2M+nb]
  const unsigned short* slotW[8] = { WCb, WCb, WHb, WHb, WDb, WDb, WHb, WHb };
  const int slotR[8] = { 0, MDIM, MDIM, 3 * MDIM, 0, MDIM, 0, 2 * MDIM };

  // Kick off staging for t = 0 into buffer 0.
  if (issuer) {
    tdm_load_tile(slotW[wave], slotR[wave] + 0,
                  lds_base + (unsigned)(wave * SLOT_U4) * 16u);
  }

  // Per-wave A operands, gathered once (pre-update values).
  FragAB a_msg[8], a_mem[8], a_dt[8];
  int idrows[8];
  if (active) {
    const int idA = (int)ids[rb + mA];
    load_a_row(a_msg, msg + (size_t)(rb + mA) * MDIM, kh);
    load_a_row(a_mem, mem + (size_t)idA * MDIM, kh);
    if (FINAL) load_a_row(a_dt, dtdg + (size_t)idA * MDIM, kh);
#pragma unroll
    for (int i = 0; i < 8; ++i) idrows[i] = (int)ids[rb + kh * 8 + i];
  }

  const int n = lane & 15;

  for (int t = 0; t < 16; ++t) {
    const int nb = t * 16;
    const int buf = t & 1;
    if (issuer) {
      if (t + 1 < 16) {   // prefetch next tile set into the other buffer
        tdm_load_tile(slotW[wave], slotR[wave] + (t + 1) * 16,
                      lds_base +
                          (unsigned)(((t + 1) & 1) * NSLOTS + wave) *
                              (unsigned)SLOT_U4 * 16u);
        __builtin_amdgcn_s_wait_tensorcnt(1);   // in-order: set t complete
      } else {
        __builtin_amdgcn_s_wait_tensorcnt(0);
      }
    }
    __syncthreads();                    // buffer `buf` ready for everyone
    if (active) {
      const int b0 = buf * NSLOTS * SLOT_U4;
      v8f iC_lo = gemm_tile_lds(a_msg, b0 + 0 * SLOT_U4, nb,            bC, lane);
      v8f iC_hi = gemm_tile_lds(a_msg, b0 + 1 * SLOT_U4, MDIM + nb,     bC, lane);
      v8f h1    = gemm_tile_lds(a_mem, b0 + 2 * SLOT_U4, MDIM + nb,     bH, lane);
      v8f h3    = gemm_tile_lds(a_mem, b0 + 3 * SLOT_U4, 3 * MDIM + nb, bH, lane);
      if (!FINAL) {
#pragma unroll
        for (int i = 0; i < 8; ++i) {
          const size_t off = (size_t)idrows[i] * MDIM + nb + n;
          float mg = mem[off];
          float g  = fsigmoid(iC_lo[i] + h1[i]);
          float hc = ftanh(iC_hi[i] + h3[i]);
          mem[off] = (1.0f - g) * hc + g * mg;
        }
      } else {
        v8f iD_lo = gemm_tile_lds(a_dt,  b0 + 4 * SLOT_U4, nb,            bD, lane);
        v8f iD_hi = gemm_tile_lds(a_dt,  b0 + 5 * SLOT_U4, MDIM + nb,     bD, lane);
        v8f h0    = gemm_tile_lds(a_mem, b0 + 6 * SLOT_U4, nb,            bH, lane);
        v8f h2    = gemm_tile_lds(a_mem, b0 + 7 * SLOT_U4, 2 * MDIM + nb, bH, lane);
#pragma unroll
        for (int i = 0; i < 8; ++i) {
          const size_t off = (size_t)idrows[i] * MDIM + nb + n;
          float mg = mem[off];
          float gD = fsigmoid(iD_lo[i] + h0[i]);
          float gC = fsigmoid(iC_lo[i] + h1[i]);
          float hD = ftanh(iD_hi[i] + h2[i]);
          float hC = ftanh(iC_hi[i] + h3[i]);
          mem[off] = 0.5f * (gD * hD + gC * hC + (2.0f - gD - gC) * mg);
        }
      }
    }
    __syncthreads();                    // everyone done reading buffer `buf`
  }
}

extern "C" void kernel_launch(void* const* d_in, const int* in_sizes, int n_in,
                              void* d_out, int out_size, void* d_ws,
                              size_t ws_size, hipStream_t stream) {
  (void)in_sizes; (void)n_in; (void)out_size; (void)ws_size;
  const float*     memory   = (const float*)d_in[0];
  const long long* node_ids = (const long long*)d_in[1];   // [8, 20000] int64
  const float*     messages = (const float*)d_in[2];       // [8, 20000, 256]
  const float*     dtdg     = (const float*)d_in[3];       // [100000, 256]
  const float*     WCw      = (const float*)d_in[4];
  const float*     bC       = (const float*)d_in[5];
  const float*     WDw      = (const float*)d_in[6];
  const float*     bD       = (const float*)d_in[7];
  const float*     WHw      = (const float*)d_in[8];
  const float*     bH       = (const float*)d_in[9];

  float* out = (float*)d_out;
  unsigned short* wbf = (unsigned short*)d_ws;

  const int lds_small = 2 * 4 * SLOT_U4 * 16;   // 67584 B (double buf, 4 slots)
  const int lds_final = 2 * 8 * SLOT_U4 * 16;   // 135168 B (double buf, 8 slots)
  static bool attr_set = false;   // host-side only; does not affect device work
  if (!attr_set) {
    (void)hipFuncSetAttribute((const void*)tgn_step<false>,
                              hipFuncAttributeMaxDynamicSharedMemorySize,
                              lds_small);
    (void)hipFuncSetAttribute((const void*)tgn_step<true>,
                              hipFuncAttributeMaxDynamicSharedMemorySize,
                              lds_final);
    attr_set = true;
  }

  // out starts as a copy of memory; all steps update it in place.
  (void)hipMemcpyAsync(out, memory, sizeof(float) * (size_t)N_NODES * MDIM,
                       hipMemcpyDeviceToDevice, stream);
  cvt_weights<<<(4 * WSZ) / 256, 256, 0, stream>>>(WCw, WDw, WHw, wbf);

  const unsigned short* WCb = wbf;
  const unsigned short* WDb = wbf + WSZ;
  const unsigned short* WHb = wbf + 2 * WSZ;

  const int blocks = (TILES_B + WAVES_PER_BLOCK - 1) / WAVES_PER_BLOCK; // 157
  for (int s = 0; s < DIVIDE - 1; ++s) {
    tgn_step<false><<<blocks, THREADS, lds_small, stream>>>(
        out, node_ids + (size_t)s * BATCH,
        messages + (size_t)s * BATCH * MDIM, dtdg,
        WCb, bC, WDb, bD, WHb, bH);
  }
  tgn_step<true><<<blocks, THREADS, lds_final, stream>>>(
      out, node_ids + (size_t)(DIVIDE - 1) * BATCH,
      messages + (size_t)(DIVIDE - 1) * BATCH * MDIM, dtdg,
      WCb, bC, WDb, bD, WHb, bH);
}
